// GIMANBackbone_62612033241213
// MI455X (gfx1250) — matrix operand order repaired and checked by reading, not compiled, not run on hardware
//
#include <hip/hip_runtime.h>
#include <hip/hip_bf16.h>

// ---------------------------------------------------------------------------
// GIMAN GNN backbone for gfx1250 (MI455X): GraphConv x3 + BN + residual + MLP
// Edge gather/scatter: f32 atomics (L2-resident, this is the roofline bound).
// GEMMs: v_wmma_f32_16x16x32_bf16 with pre-converted bf16 operands so the
// inner loop is pure b128 loads + WMMA (no per-element cvt VALU).
// ---------------------------------------------------------------------------

typedef __attribute__((ext_vector_type(16))) __bf16 v16bf;
typedef __attribute__((ext_vector_type(8)))  float  v8f;
typedef unsigned short u16;

#define TPB 256
#define WAVES_PER_BLOCK 4

static __device__ __forceinline__ unsigned bfbits(float f) {
    union { float f; unsigned u; } v; v.f = f;
    return (v.u + 0x7FFFu + ((v.u >> 16) & 1u)) >> 16;   // RNE truncate to bf16
}

union V16U { uint4 q[2]; v16bf v; };

// A fragment: halves [0..7] and [16..23] from base (ISA 16-bit A 16x32 layout)
static __device__ __forceinline__ v16bf load_frag_a(const u16* __restrict__ base) {
    V16U u;
    u.q[0] = *(const uint4*)(base);
    u.q[1] = *(const uint4*)(base + 16);
    return u.v;
}
// B fragment: 16 consecutive halves from base (contiguous k = 16h + s; k_gemm uses it for both A and B)
static __device__ __forceinline__ v16bf load_frag_b(const u16* __restrict__ base) {
    V16U u;
    u.q[0] = *(const uint4*)(base);
    u.q[1] = *(const uint4*)(base + 8);
    return u.v;
}

// ---------------- zero fill ----------------
__global__ void k_zero(float* __restrict__ p, long n) {
    long i = (long)blockIdx.x * blockDim.x + threadIdx.x;
    const long stride = (long)gridDim.x * blockDim.x;
    for (; i < n; i += stride) p[i] = 0.0f;
}

// ---------------- f32 -> bf16 bulk convert (n4 = count/4) ----------------
__global__ void k_cvt(const float* __restrict__ in, u16* __restrict__ outp, long n4) {
    long i = (long)blockIdx.x * blockDim.x + threadIdx.x;
    const long stride = (long)gridDim.x * blockDim.x;
    for (; i < n4; i += stride) {
        const float4 v = ((const float4*)in)[i];
        uint2 w;
        w.x = bfbits(v.x) | (bfbits(v.y) << 16);
        w.y = bfbits(v.z) | (bfbits(v.w) << 16);
        ((uint2*)outp)[i] = w;
    }
}

// ---------------- edge scatter-add: agg[dst] += x[src] * w ----------------
__global__ void k_scatter(const float* __restrict__ x, const int* __restrict__ src,
                          const int* __restrict__ dst, const float* __restrict__ ew,
                          float* __restrict__ agg, int E, int d) {
    const int chunks = d >> 2;
    const long total = (long)E * chunks;
    long i = (long)blockIdx.x * blockDim.x + threadIdx.x;
    const long stride = (long)gridDim.x * blockDim.x;
    for (; i < total; i += stride) {
        const int e = (int)(i / chunks);
        const int j = ((int)(i % chunks)) << 2;
        const int s = src[e];
        const int t = dst[e];
        const float w = ew ? ew[e] : 1.0f;
        const float4 v = *(const float4*)(x + (long)s * d + j);
        float* a = agg + (long)t * d + j;
        atomicAdd(a + 0, v.x * w);
        atomicAdd(a + 1, v.y * w);
        atomicAdd(a + 2, v.z * w);
        atomicAdd(a + 3, v.w * w);
    }
}

// ---------------- WMMA dual GEMM (bf16 operands, f32 accumulate) ----------------
// out[M,dN] = A1[M,DK] @ W1[dN,DK]^T (+ A2 @ W2^T) + bias (+ optional ReLU)
template <int DK>
__global__ void k_gemm(const u16* __restrict__ A1, const u16* __restrict__ W1,
                       const u16* __restrict__ A2, const u16* __restrict__ W2,
                       const float* __restrict__ bias, float* __restrict__ out,
                       int n_rows, int dN, int do_relu) {
    const int tiles_n = dN >> 4;
    const int tiles_m = n_rows >> 4;     // n_rows multiple of 16 (100000 = 6250*16)
    const int wave = blockIdx.x * WAVES_PER_BLOCK + (threadIdx.x >> 5);
    if (wave >= tiles_m * tiles_n) return;   // wave-uniform: EXEC all-1s for WMMA
    const int lane = threadIdx.x & 31;
    const int m0 = (wave / tiles_n) << 4;
    const int n0 = (wave % tiles_n) << 4;
    const int ml = lane & 15;
    const int hi = lane >> 4;

    const long a_off = (long)(m0 + ml) * DK + (hi << 4);
    const long w_off = (long)(n0 + ml) * DK + (hi << 4);

    v8f c = {};
#pragma unroll
    for (int k0 = 0; k0 < DK; k0 += 32) {
        const v16bf a = load_frag_b(A1 + a_off + k0);
        const v16bf b = load_frag_b(W1 + w_off + k0);
        c = __builtin_amdgcn_wmma_f32_16x16x32_bf16(false, a, false, b,
                                                    (short)0, c, false, false);
    }
    if (A2) {
#pragma unroll
        for (int k0 = 0; k0 < DK; k0 += 32) {
            const v16bf a = load_frag_b(A2 + a_off + k0);
            const v16bf b = load_frag_b(W2 + w_off + k0);
            c = __builtin_amdgcn_wmma_f32_16x16x32_bf16(false, a, false, b,
                                                        (short)0, c, false, false);
        }
    }

    const float bv = bias ? bias[n0 + ml] : 0.0f;
    const int mbase = m0 + (hi << 3);    // C/D: VGPR r -> row m0+r / m0+8+r
#pragma unroll
    for (int r = 0; r < 8; ++r) {
        float v = c[r] + bv;
        if (do_relu) v = fmaxf(v, 0.0f);
        out[(long)(mbase + r) * dN + (n0 + ml)] = v;
    }
}

// ---------------- BN statistics: one feature column per thread ----------------
__global__ void k_bnstats(const float* __restrict__ h, float* __restrict__ sum,
                          float* __restrict__ sumsq, int n_rows, int d, int rows_per_block) {
    const int j = threadIdx.x;
    const int r0 = blockIdx.x * rows_per_block;
    int r1 = r0 + rows_per_block;
    if (r1 > n_rows) r1 = n_rows;
    float s = 0.0f, s2 = 0.0f;
    for (int r = r0; r < r1; ++r) {
        const float v = h[(long)r * d + j];
        s += v;
        s2 += v * v;
    }
    atomicAdd(&sum[j], s);
    atomicAdd(&sumsq[j], s2);
}

__global__ void k_bnfinal(const float* __restrict__ sum, const float* __restrict__ sumsq,
                          const float* __restrict__ g, const float* __restrict__ be,
                          float* __restrict__ scale, float* __restrict__ shift,
                          int d, float inv_n) {
    const int j = blockIdx.x * blockDim.x + threadIdx.x;
    if (j >= d) return;
    const float mu = sum[j] * inv_n;
    const float var = sumsq[j] * inv_n - mu * mu;
    const float sc = g[j] * rsqrtf(var + 1e-5f);
    scale[j] = sc;
    shift[j] = be[j] - mu * sc;
}

// ------- BN apply (+residual, ReLU) in place; also writes bf16 mirror -------
__global__ void k_bnapply(float* __restrict__ h, const float* __restrict__ scale,
                          const float* __restrict__ shift, const float* __restrict__ res,
                          u16* __restrict__ hbf, long total, int d, int do_relu) {
    long i = (long)blockIdx.x * blockDim.x + threadIdx.x;
    const long stride = (long)gridDim.x * blockDim.x;
    for (; i < total; i += stride) {
        const int j = (int)(i % d);
        float v = h[i] * scale[j] + shift[j];
        if (res) v += res[i];
        if (do_relu) v = fmaxf(v, 0.0f);
        h[i] = v;
        hbf[i] = (u16)bfbits(v);
    }
}

// -------- classifier head: out[n,0:2] = hc[n,:32] @ Wc2^T + bc2 --------
__global__ void k_head(const float* __restrict__ hc, const float* __restrict__ W2,
                       const float* __restrict__ b2, float* __restrict__ out, int n_rows) {
    const int n = blockIdx.x * blockDim.x + threadIdx.x;
    if (n >= n_rows) return;
    const float* row = hc + (long)n * 32;
    float a0 = b2[0], a1 = b2[1];
#pragma unroll
    for (int k = 0; k < 32; ++k) {
        const float v = row[k];
        a0 += v * W2[k];
        a1 += v * W2[32 + k];
    }
    out[(long)n * 2 + 0] = a0;
    out[(long)n * 2 + 1] = a1;
}

// ---------------------------------------------------------------------------
static inline int cdiv(long a, long b) { return (int)((a + b - 1) / b); }

extern "C" void kernel_launch(void* const* d_in, const int* in_sizes, int n_in,
                              void* d_out, int out_size, void* d_ws, size_t ws_size,
                              hipStream_t stream) {
    const float* x    = (const float*)d_in[0];
    const int*   ei   = (const int*)d_in[1];
    const float* ew   = (const float*)d_in[2];
    const float* W1r  = (const float*)d_in[3];
    const float* b1   = (const float*)d_in[4];
    const float* W1o  = (const float*)d_in[5];
    const float* W2r  = (const float*)d_in[6];
    const float* b2   = (const float*)d_in[7];
    const float* W2o  = (const float*)d_in[8];
    const float* W3r  = (const float*)d_in[9];
    const float* b3   = (const float*)d_in[10];
    const float* W3o  = (const float*)d_in[11];
    const float* g1   = (const float*)d_in[12];
    const float* be1  = (const float*)d_in[13];
    const float* g2   = (const float*)d_in[14];
    const float* be2  = (const float*)d_in[15];
    const float* g3   = (const float*)d_in[16];
    const float* be3  = (const float*)d_in[17];
    const float* Wc1  = (const float*)d_in[18];
    const float* bc1  = (const float*)d_in[19];
    const float* Wc2  = (const float*)d_in[20];
    const float* bc2  = (const float*)d_in[21];

    const int N = in_sizes[0] / 32;      // 100000
    const int E = in_sizes[2];           // 1600000
    const int* src = ei;
    const int* dst = ei + E;

    // -------- workspace layout --------
    float* ws    = (float*)d_ws;
    float* agg   = ws;                        // N*128 f32 (also classifier hidden)
    float* h1    = agg + (size_t)N * 128;     // N*64
    float* h2    = h1  + (size_t)N * 64;      // N*128
    float* h3    = h2  + (size_t)N * 128;     // N*64
    float* stats = h3  + (size_t)N * 64;      // 512
    float* ssum   = stats;
    float* ssq    = stats + 128;
    float* sscale = stats + 256;
    float* sshift = stats + 384;
    u16* abf = (u16*)(stats + 512);           // N*128 bf16 : agg mirror
    u16* xbf = abf + (size_t)N * 128;         // N*128 bf16 : activation mirror
    u16* wbf = xbf + (size_t)N * 128;         // weight mirrors (bf16)
    u16* W1r_bf = wbf;                 // 64*32
    u16* W1o_bf = W1r_bf + 2048;       // 64*32
    u16* W2r_bf = W1o_bf + 2048;       // 128*64
    u16* W2o_bf = W2r_bf + 8192;       // 128*64
    u16* W3r_bf = W2o_bf + 8192;       // 64*128
    u16* W3o_bf = W3r_bf + 8192;       // 64*128
    u16* Wc1_bf = W3o_bf + 8192;       // 32*64

    // -------- one-time conversions: weights + input x --------
    k_cvt<<<cdiv(2048 / 4, TPB), TPB, 0, stream>>>(W1r, W1r_bf, 2048 / 4);
    k_cvt<<<cdiv(2048 / 4, TPB), TPB, 0, stream>>>(W1o, W1o_bf, 2048 / 4);
    k_cvt<<<cdiv(8192 / 4, TPB), TPB, 0, stream>>>(W2r, W2r_bf, 8192 / 4);
    k_cvt<<<cdiv(8192 / 4, TPB), TPB, 0, stream>>>(W2o, W2o_bf, 8192 / 4);
    k_cvt<<<cdiv(8192 / 4, TPB), TPB, 0, stream>>>(W3r, W3r_bf, 8192 / 4);
    k_cvt<<<cdiv(8192 / 4, TPB), TPB, 0, stream>>>(W3o, W3o_bf, 8192 / 4);
    k_cvt<<<cdiv(2048 / 4, TPB), TPB, 0, stream>>>(Wc1, Wc1_bf, 2048 / 4);
    k_cvt<<<4096, TPB, 0, stream>>>(x, xbf, (long)N * 32 / 4);   // layer-1 root input

    struct Layer {
        const float* xin; int din, dout;
        const u16* Wrel_bf; const float* bias; const u16* Wroot_bf;
        const float* g; const float* be; const float* ewt;
        const float* resid; float* hout;
    };
    const Layer layers[3] = {
        { x,  32,  64, W1r_bf, b1, W1o_bf, g1, be1, nullptr, nullptr, h1 },
        { h1, 64, 128, W2r_bf, b2, W2o_bf, g2, be2, nullptr, nullptr, h2 },
        { h2, 128, 64, W3r_bf, b3, W3o_bf, g3, be3, ew,      h1,      h3 },
    };

    for (int L = 0; L < 3; ++L) {
        const Layer& ly = layers[L];
        const long agg_n = (long)N * ly.din;

        // 1. zero aggregation buffer, 2. edge scatter-add (L2 atomics)
        k_zero<<<cdiv(agg_n, TPB), TPB, 0, stream>>>(agg, agg_n);
        const long sc_threads = (long)E * (ly.din >> 2);
        k_scatter<<<cdiv(sc_threads, TPB), TPB, 0, stream>>>(
            ly.xin, src, dst, ly.ewt, agg, E, ly.din);
        // 3. agg -> bf16 mirror
        k_cvt<<<4096, TPB, 0, stream>>>(agg, abf, agg_n / 4);
        // 4. dual WMMA GEMM: agg@Wrel^T + xin@Wroot^T + b
        const int tiles = (N >> 4) * (ly.dout >> 4);
        const int gblk = cdiv(tiles, WAVES_PER_BLOCK);
        if (ly.din == 32)
            k_gemm<32><<<gblk, 32 * WAVES_PER_BLOCK, 0, stream>>>(
                abf, ly.Wrel_bf, xbf, ly.Wroot_bf, ly.bias, ly.hout, N, ly.dout, 0);
        else if (ly.din == 64)
            k_gemm<64><<<gblk, 32 * WAVES_PER_BLOCK, 0, stream>>>(
                abf, ly.Wrel_bf, xbf, ly.Wroot_bf, ly.bias, ly.hout, N, ly.dout, 0);
        else
            k_gemm<128><<<gblk, 32 * WAVES_PER_BLOCK, 0, stream>>>(
                abf, ly.Wrel_bf, xbf, ly.Wroot_bf, ly.bias, ly.hout, N, ly.dout, 0);
        // 5. BatchNorm stats -> finalize -> apply (+residual, ReLU, bf16 mirror)
        k_zero<<<1, TPB, 0, stream>>>(stats, 256);
        k_bnstats<<<cdiv(N, 256), ly.dout, 0, stream>>>(ly.hout, ssum, ssq, N, ly.dout, 256);
        k_bnfinal<<<1, 128, 0, stream>>>(ssum, ssq, ly.g, ly.be, sscale, sshift,
                                         ly.dout, 1.0f / (float)N);
        const long hn = (long)N * ly.dout;
        const int ablk = cdiv(hn, TPB) < 4096 ? cdiv(hn, TPB) : 4096;
        k_bnapply<<<ablk, TPB, 0, stream>>>(ly.hout, sscale, sshift, ly.resid,
                                            xbf, hn, ly.dout, 1);
    }

    // classifier: hc = relu(h3 @ Wc1^T + bc1)  [N,32]  (h3 bf16 mirror is in xbf)
    float* hc = agg;
    const int tiles_c = (N >> 4) * 2;
    k_gemm<64><<<cdiv(tiles_c, WAVES_PER_BLOCK), 32 * WAVES_PER_BLOCK, 0, stream>>>(
        xbf, Wc1_bf, nullptr, nullptr, bc1, hc, N, 32, 1);
    // logits = hc @ Wc2^T + bc2  [N,2]
    k_head<<<cdiv(N, TPB), TPB, 0, stream>>>(hc, Wc2, bc2, (float*)d_out, N);
}
